// PyUriTwc_V2_24309514895624
// MI455X (gfx1250) — matrix operand order repaired and checked by reading, not compiled, not run on hardware
//
#include <hip/hip_runtime.h>
#include <hip/hip_bf16.h>

#define B_    16
#define T_    32
#define N_    512
#define NIN_  32
#define NOUT_ 16

typedef _Float16 v16h __attribute__((ext_vector_type(16)));
typedef _Float16 h8   __attribute__((ext_vector_type(8)));
typedef float    v8f  __attribute__((ext_vector_type(8)));

__device__ __forceinline__ float softplus_f(float x) {
    // numerically stable softplus
    return fmaxf(x, 0.0f) + log1pf(expf(-fabsf(x)));
}

__device__ __forceinline__ float fast_tanh10(float diff) {
    // tanh(10*diff) = 1 - 2/(exp2(10*diff*2*log2(e)) + 1)
    float e = __builtin_amdgcn_exp2f(diff * 28.853900817779268f);
    return 1.0f - 2.0f * __builtin_amdgcn_rcpf(e + 1.0f);
}

__device__ __forceinline__ float fast_sigmoid(float y) {
    // 1 / (1 + exp(-y));  exp(-y) = exp2(-y*log2(e))
    return __builtin_amdgcn_rcpf(1.0f + __builtin_amdgcn_exp2f(-y * 1.4426950408889634f));
}

// ---------------------------------------------------------------------------
// Precompute 1: W_chem packed into a WMMA f16 B-operand layout with contiguous k order.
// B (32x16 f16) as packed here: VGPR v, lanes 0-15 hold K=2v,2v+1 (col = lane); lanes 16-31
// hold K=16+2v,16+2v+1.  Flattened as v16h element e -> K offset = e, with
// +16 for the upper half-wave.  Buffer layout:
//   packed[(((tile*16 + kchunk)*32) + lane)*16 + e]
//   where d = tile*16 + (lane&15), s = kchunk*32 + ((lane&16)?16:0) + e
// ---------------------------------------------------------------------------
__global__ void pack_chem_kernel(const float* __restrict__ W,
                                 const float* __restrict__ mex,
                                 const float* __restrict__ min_,
                                 _Float16* __restrict__ packed) {
    int i = blockIdx.x * blockDim.x + threadIdx.x;
    if (i >= N_ * N_) return;
    int e    = i & 15;
    int lane = (i >> 4) & 31;
    int kc   = (i >> 9) & 15;
    int tl   = i >> 13;
    int d = tl * 16 + (lane & 15);
    int s = kc * 32 + ((lane & 16) ? 16 : 0) + e;
    float w = softplus_f(W[d * N_ + s]) * (mex[d * N_ + s] - min_[d * N_ + s]);
    packed[i] = (_Float16)w;
}

// Precompute 2: W_gj transposed (f32, [s][d]) so the gj inner loop is
// lane-coalesced in d.
__global__ void make_gjT_kernel(const float* __restrict__ W,
                                const float* __restrict__ mgj,
                                float* __restrict__ wgjT) {
    int i = blockIdx.x * blockDim.x + threadIdx.x;
    if (i >= N_ * N_) return;
    int s = i >> 9;       // i = s*N + d
    int d = i & (N_ - 1);
    wgjT[i] = softplus_f(W[d * N_ + s]) * mgj[d * N_ + s];
}

// ---------------------------------------------------------------------------
// Main scan: one workgroup per batch (batches are independent -> no global
// sync).  1024 threads = 32 wave32s on one WGP.  State (E,O) lives in LDS.
//   - chem term: each wave owns a 16-wide d-tile; v_wmma_f32_16x16x32_f16
//     with the O-vector replicated across the 16 M rows (row 0 of D is the
//     GEMV result).  Runs on the matrix pipe, co-executing with the VALU.
//     The 512KB packed W_chem stays L2-resident; B tiles are re-loaded from
//     L2 every step via GLOBAL loads (opaque *offset* blocks LICM without
//     losing the global address space -> no flat_load, no scratch spills).
//   - gj term: 2 threads per neuron split the s-range; tanh via v_exp/v_rcp.
// ---------------------------------------------------------------------------
__global__ void __launch_bounds__(1024, 1)
rnn_scan_kernel(const float* __restrict__ obs,
                const float* __restrict__ thr,
                const float* __restrict__ dec,
                const int*   __restrict__ inIdx,
                const int*   __restrict__ outIdx,
                const _Float16* __restrict__ wchemP,
                const float* __restrict__ wgjT,
                float* __restrict__ out) {
    __shared__ float Esh[N_];
    __shared__ float Osh[N_];
    __shared__ __align__(16) _Float16 Oh16[N_];
    __shared__ float chem[N_];
    __shared__ float gj2[1024];

    const int tid  = threadIdx.x;
    const int b    = blockIdx.x;
    const int lane = tid & 31;
    const int w    = tid >> 5;          // wave id 0..31 -> d tile

    if (tid < N_) { Esh[tid] = 0.0f; Osh[tid] = 0.0f; Oh16[tid] = (_Float16)0.0f; }
    __syncthreads();

    const int half = tid >> 9;          // 0 or 1: s-range split for gj
    const int d    = tid & (N_ - 1);
    const int s0   = half * (N_ / 2);
    const int laneHiA = (lane & 16) ? 16 : 0;  // A-operand K offset for upper half-wave
    const float thr_d = thr[d];
    const float dec_d = dec[d];
    const _Float16* bp0 = wchemP + ((size_t)(w * 16) * 32 + lane) * 16;

    for (int t = 0; t < T_; ++t) {
        // --- inject observations into state (input_indices) ---
        if (tid < NIN_) {
            int ii  = inIdx[tid];
            float v = obs[(b * T_ + t) * NIN_ + tid];
            Osh[ii] = v; Esh[ii] = v; Oh16[ii] = (_Float16)v;
        }
        __syncthreads();

        // --- chem = O_h @ W_chem^T for this wave's 16-wide d tile (WMMA) ---
        v8f acc = {};
        // Opaque 32-bit offset: forbids hoisting the 16 B-tile loads (128
        // VGPRs of loop-invariant data) out of the t-loop (which forced
        // scratch spills), while keeping bp derived from the kernel-arg
        // pointer so the loads stay GLOBAL (LOADcnt only), not FLAT.
        unsigned kopq = 0;
        asm volatile("" : "+v"(kopq));
        const _Float16* bp = bp0 + kopq;
        #pragma unroll
        for (int kc = 0; kc < 16; ++kc) {
            // A: replicated O rows; two 16B LDS loads per lane
            h8 alo = *(const h8*)&Oh16[kc * 32 + laneHiA];
            h8 ahi = *(const h8*)&Oh16[kc * 32 + laneHiA + 8];
            v16h a = __builtin_shufflevector(alo, ahi,
                      0, 1, 2, 3, 4, 5, 6, 7, 8, 9, 10, 11, 12, 13, 14, 15);
            v16h bm = *(const v16h*)bp;
            __builtin_prefetch(bp + 512, 0, 1);   // global_prefetch_b8: next K chunk
            acc = __builtin_amdgcn_wmma_f32_16x16x32_f16(
                      false, a, false, bm, (short)0, acc, false, false);
            bp += 512;
        }
        if (lane < 16) chem[w * 16 + lane] = acc[0];   // D row M=0 = GEMV result

        // --- gj partial sums: thread (d, half) covers 256 s values ---
        const float Ed = Esh[d];
        float acc_gj = 0.0f;
        #pragma unroll 4
        for (int s = s0; s < s0 + N_ / 2; ++s) {
            float o  = Osh[s];
            float wg = wgjT[s * N_ + d];          // lane-coalesced, L2 resident
            float th = fast_tanh10(o - Ed);
            acc_gj = fmaf(wg * o, th, acc_gj);
        }
        gj2[tid] = acc_gj;
        __syncthreads();

        // --- state update (one thread per neuron) ---
        if (tid < N_) {
            float I    = chem[d] + gj2[d] + gj2[d + N_];
            float curr = fminf(fmaxf(Ed + I, -10.0f), 10.0f);
            float z    = curr - thr_d;
            float on   = (z >= 0.0f) ? z : 0.01f * z;
            float fg   = fast_sigmoid(10.0f * z);
            float dg   = fast_sigmoid(5.0f * (fabsf(Ed - curr) - 0.01f));
            float enf  = dg * curr + (1.0f - dg) * (Ed - dec_d);
            float en   = fg * on + (1.0f - fg) * enf;
            Osh[d] = on; Esh[d] = en; Oh16[d] = (_Float16)on;
        }
        __syncthreads();

        // --- emit actions ---
        if (tid < NOUT_) {
            out[(b * T_ + t) * NOUT_ + tid] = Esh[outIdx[tid]];
        }
        __syncthreads();
    }
}

extern "C" void kernel_launch(void* const* d_in, const int* in_sizes, int n_in,
                              void* d_out, int out_size, void* d_ws, size_t ws_size,
                              hipStream_t stream) {
    const float* obs    = (const float*)d_in[0];
    const float* W      = (const float*)d_in[1];
    const float* thr    = (const float*)d_in[2];
    const float* dec    = (const float*)d_in[3];
    const float* mex    = (const float*)d_in[4];
    const float* min_   = (const float*)d_in[5];
    const float* mgj    = (const float*)d_in[6];
    const int*   inIdx  = (const int*)d_in[7];
    const int*   outIdx = (const int*)d_in[8];
    float*       out    = (float*)d_out;

    // workspace: [0, 512KB) packed f16 W_chem tiles; [512KB, 1.5MB) f32 W_gj^T
    _Float16* wchemP = (_Float16*)d_ws;
    float*    wgjT   = (float*)((char*)d_ws + (size_t)N_ * N_ * sizeof(_Float16));

    const int total = N_ * N_;
    pack_chem_kernel<<<total / 256, 256, 0, stream>>>(W, mex, min_, wchemP);
    make_gjT_kernel<<<total / 256, 256, 0, stream>>>(W, mgj, wgjT);
    rnn_scan_kernel<<<B_, 1024, 0, stream>>>(obs, thr, dec, inIdx, outIdx,
                                             wchemP, wgjT, out);
}
